// RowParallelLinearWithDelta_79766132621353
// MI455X (gfx1250) — compile-verified, matrix operand order checked by reading, not run on hardware
//
#include <hip/hip_runtime.h>
#include <hip/hip_bf16.h>

// ---------------------------------------------------------------------------
// RowParallelLinearWithDelta for MI455X (gfx1250, wave32, WMMA)
//   out = x @ W^T + bias + per-token int4 LoRA delta (8 adapters, 1 group)
// Strategy: token grouping -> exactly 2 bf16 WMMA GEMMs of 4096^3.
// K-loop is software-pipelined: next tile's global loads are issued into
// registers while the current tile's WMMAs run from LDS.
// ---------------------------------------------------------------------------

typedef __bf16 bf16_t;
typedef __attribute__((ext_vector_type(16))) __bf16 v16bf;
typedef __attribute__((ext_vector_type(8)))  __bf16 v8bf;
typedef __attribute__((ext_vector_type(8)))  float  v8f;

#define T_TOK   4096
#define IN_DIM  4096
#define OUT_DIM 4096
#define N_ADPT  8

#define BM 128
#define BN 128
#define BK 32
#define KS 40   // LDS row stride in bf16 elements (80B: 16B-aligned, bank-skewed)

// Load one 16x16x32 bf16 A/B fragment from an LDS tile laid out [row][K],
// following the CDNA5 VGPR layout: lanes 0-15 hold row r0+lane with K
// {0..7,16..23}; lanes 16-31 hold the same rows with K {8..15,24..31}.
__device__ inline v16bf frag_from_lds(const bf16_t* tile, int r0, int lane) {
    const int r  = r0 + (lane & 15);
    const int ks = (lane >> 4) << 3;            // 0 or 8
    const bf16_t* p = tile + r * KS + ks;
    v8bf lo = *(const v8bf*)(p);                // K = ks .. ks+7
    v8bf hi = *(const v8bf*)(p + 16);           // K = ks+16 .. ks+23
    v16bf f;
#pragma unroll
    for (int i = 0; i < 8; ++i) { f[i] = lo[i]; f[i + 8] = hi[i]; }
    return f;
}

__device__ inline void cvt16(const float4* r, bf16_t* t) {
#pragma unroll
    for (int i = 0; i < 4; ++i) {
        t[i*4+0] = (bf16_t)r[i].x; t[i*4+1] = (bf16_t)r[i].y;
        t[i*4+2] = (bf16_t)r[i].z; t[i*4+3] = (bf16_t)r[i].w;
    }
}

// ---------------------------------------------------------------------------
// Kernel 1: group tokens by adapter index. offsets[9] + perm[T] into d_ws.
// ---------------------------------------------------------------------------
__global__ __launch_bounds__(256) void group_kernel(const int* __restrict__ indices,
                                                    int* __restrict__ offsets,
                                                    int* __restrict__ perm) {
    __shared__ int cnt[N_ADPT], base[N_ADPT], cur[N_ADPT];
    const int tid = threadIdx.x;
    if (tid < N_ADPT) { cnt[tid] = 0; cur[tid] = 0; }
    __syncthreads();
    for (int t = tid; t < T_TOK; t += 256)
        atomicAdd(&cnt[indices[t]], 1);
    __syncthreads();
    if (tid == 0) {
        int s = 0;
        for (int d = 0; d < N_ADPT; ++d) { base[d] = s; offsets[d] = s; s += cnt[d]; }
        offsets[N_ADPT] = s;
    }
    __syncthreads();
    for (int t = tid; t < T_TOK; t += 256) {
        int d = indices[t];
        int pos = base[d] + atomicAdd(&cur[d], 1);
        perm[pos] = t;   // slot order is racy, but each row computes independently
    }
}

// ---------------------------------------------------------------------------
// Kernel 2: base GEMM  out = x @ W^T + bias   (f32 -> bf16 WMMA, f32 accum)
// Block: 256 thr (8 waves), tile 128x128, K-step 32, register double-buffer.
// Wave (wm,wn) in 4x2 grid owns 32(M) x 64(N): acc[2][4] of v8f.
// ---------------------------------------------------------------------------
__global__ __launch_bounds__(256) void base_gemm(const float* __restrict__ x,
                                                 const float* __restrict__ w,
                                                 const float* __restrict__ bias,
                                                 float* __restrict__ out) {
    __shared__ __align__(16) bf16_t As[BM * KS];
    __shared__ __align__(16) bf16_t Bs[BN * KS];

    const int tid  = threadIdx.x;
    const int lane = tid & 31, wave = tid >> 5;
    const int wm = wave >> 1, wn = wave & 1;
    const int m0 = blockIdx.y * BM, n0 = blockIdx.x * BN;

    v8f acc[2][4] = {};

    const int lrow = tid >> 1;              // 0..127: tile row
    const int lseg = (tid & 1) << 4;        // 0 or 16: K segment

    const float* srcA = x + (size_t)(m0 + lrow) * IN_DIM + lseg;
    const float* srcB = w + (size_t)(n0 + lrow) * IN_DIM + lseg;
    bf16_t* ldsA = &As[lrow * KS + lseg];
    bf16_t* ldsB = &Bs[lrow * KS + lseg];

    // prefetch tile 0
    float4 ra[4], rb[4];
#pragma unroll
    for (int i = 0; i < 4; ++i) {
        ra[i] = *(const float4*)(srcA + i * 4);
        rb[i] = *(const float4*)(srcB + i * 4);
    }
    srcA += BK; srcB += BK;

    for (int k0 = 0; k0 < IN_DIM; k0 += BK) {
        __syncthreads();               // previous compute done reading LDS
        {
            alignas(16) bf16_t ta[16], tb[16];
            cvt16(ra, ta); cvt16(rb, tb);
            *(v8bf*)(ldsA)     = *(v8bf*)&ta[0];
            *(v8bf*)(ldsA + 8) = *(v8bf*)&ta[8];
            *(v8bf*)(ldsB)     = *(v8bf*)&tb[0];
            *(v8bf*)(ldsB + 8) = *(v8bf*)&tb[8];
        }
        __syncthreads();
        if (k0 + BK < IN_DIM) {        // prefetch next tile (overlaps compute)
#pragma unroll
            for (int i = 0; i < 4; ++i) {
                ra[i] = *(const float4*)(srcA + i * 4);
                rb[i] = *(const float4*)(srcB + i * 4);
            }
            srcA += BK; srcB += BK;
        }

        v16bf a0 = frag_from_lds(As, wm * 32,      lane);
        v16bf a1 = frag_from_lds(As, wm * 32 + 16, lane);
        v16bf b[4];
#pragma unroll
        for (int j = 0; j < 4; ++j) b[j] = frag_from_lds(Bs, wn * 64 + j * 16, lane);
#pragma unroll
        for (int j = 0; j < 4; ++j) {
            acc[0][j] = __builtin_amdgcn_wmma_f32_16x16x32_bf16(false, a0, false, b[j],
                                                                (short)0, acc[0][j], false, false);
            acc[1][j] = __builtin_amdgcn_wmma_f32_16x16x32_bf16(false, a1, false, b[j],
                                                                (short)0, acc[1][j], false, false);
        }
    }

    // Epilogue: C layout -> lane%16 = col, VGPR v = row v (+8 if lane>=16)
    const int colBase = n0 + wn * 64 + (lane & 15);
    const int rowHalf = (lane >> 4) * 8;
#pragma unroll
    for (int i = 0; i < 2; ++i) {
#pragma unroll
        for (int j = 0; j < 4; ++j) {
            const int col = colBase + j * 16;
            const float bv = bias[col];
#pragma unroll
            for (int v = 0; v < 8; ++v) {
                const int row = m0 + wm * 32 + i * 16 + rowHalf + v;
                out[(size_t)row * OUT_DIM + col] = acc[i][j][v] + bv;
            }
        }
    }
}

// ---------------------------------------------------------------------------
// Kernel 3: per-adapter delta GEMM with gathered token rows and int4->bf16
// dequant fused into the B-tile LDS loader.  out[t][o] += (x[t] @ Wd)[o]
// ---------------------------------------------------------------------------
__global__ __launch_bounds__(256) void delta_gemm(const float* __restrict__ x,
                                                  const int* __restrict__ qw,
                                                  const int* __restrict__ qz,
                                                  const float* __restrict__ scales,
                                                  const int* __restrict__ offsets,
                                                  const int* __restrict__ perm,
                                                  float* __restrict__ out) {
    const int d      = blockIdx.z;
    const int gstart = offsets[d];
    const int cnt    = offsets[d + 1] - gstart;
    const int m0     = blockIdx.y * BM;
    if (m0 >= cnt) return;                      // uniform across block
    const int n0 = blockIdx.x * BN;

    __shared__ __align__(16) bf16_t As[BM * KS];
    __shared__ __align__(16) bf16_t Bs[BN * KS];
    __shared__ float sc[BN];
    __shared__ float zz[BN];
    __shared__ int   tok[BM];

    const int tid  = threadIdx.x;
    const int lane = tid & 31, wave = tid >> 5;
    const int wm = wave >> 1, wn = wave & 1;

    if (tid < BM) {
        const int idx = m0 + tid;
        tok[tid] = (idx < cnt) ? perm[gstart + idx] : -1;
    }
    if (tid < BN) {
        const int gcol = n0 + tid;
        sc[tid] = scales[(size_t)d * OUT_DIM + gcol];
        const int z = qz[(size_t)d * (OUT_DIM / 8) + (gcol >> 3)];
        zz[tid] = (float)((z >> ((gcol & 7) * 4)) & 0xF);
    }
    __syncthreads();                            // tok/sc/zz visible

    v8f acc[2][4] = {};

    const int lrow = tid >> 1;
    const int lseg = (tid & 1) << 4;
    const int myTok = tok[lrow];
    const float* srcA = (myTok >= 0) ? x + (size_t)myTok * IN_DIM + lseg : x;
    bf16_t* ldsA = &As[lrow * KS + lseg];

    // B loader geometry: unit u covers (col o_u, packed-K row kr_u); each
    // int32 of qweight packs 8 consecutive K values at one output column.
    const size_t qbase = (size_t)d * (IN_DIM / 8) * OUT_DIM;
    const int o0  = tid & (BN - 1),        o1  = o0;
    const int kr0 = tid >> 7,              kr1 = kr0 + 2;
    const int* srcQ0 = qw + qbase + (size_t)kr0 * OUT_DIM + n0 + o0;
    const int* srcQ1 = qw + qbase + (size_t)kr1 * OUT_DIM + n0 + o1;
    const float s0 = sc[o0], z0 = zz[o0];

    // prefetch tile 0
    float4 ra[4]; int rq0, rq1;
    if (myTok >= 0) {
#pragma unroll
        for (int i = 0; i < 4; ++i) ra[i] = *(const float4*)(srcA + i * 4);
    }
    rq0 = *srcQ0; rq1 = *srcQ1;
    srcA += BK; srcQ0 += (BK / 8) * OUT_DIM; srcQ1 += (BK / 8) * OUT_DIM;

    for (int k0 = 0; k0 < IN_DIM; k0 += BK) {
        __syncthreads();
        // A tile store (zeros for padded slots)
        {
            alignas(16) bf16_t ta[16];
            if (myTok >= 0) cvt16(ra, ta);
            else {
#pragma unroll
                for (int i = 0; i < 16; ++i) ta[i] = (bf16_t)0.0f;
            }
            *(v8bf*)(ldsA)     = *(v8bf*)&ta[0];
            *(v8bf*)(ldsA + 8) = *(v8bf*)&ta[8];
        }
        // B tile store: dequantize int4 -> bf16
        {
            alignas(16) bf16_t v0[8], v1[8];
#pragma unroll
            for (int j = 0; j < 8; ++j) {
                v0[j] = (bf16_t)(s0 * ((float)((rq0 >> (j * 4)) & 0xF) - z0));
                v1[j] = (bf16_t)(s0 * ((float)((rq1 >> (j * 4)) & 0xF) - z0));
            }
            *(v8bf*)&Bs[o0 * KS + kr0 * 8] = *(v8bf*)v0;
            *(v8bf*)&Bs[o1 * KS + kr1 * 8] = *(v8bf*)v1;
        }
        __syncthreads();
        if (k0 + BK < IN_DIM) {        // prefetch next tile (overlaps compute)
            if (myTok >= 0) {
#pragma unroll
                for (int i = 0; i < 4; ++i) ra[i] = *(const float4*)(srcA + i * 4);
            }
            rq0 = *srcQ0; rq1 = *srcQ1;
            srcA += BK; srcQ0 += (BK / 8) * OUT_DIM; srcQ1 += (BK / 8) * OUT_DIM;
        }

        v16bf a0 = frag_from_lds(As, wm * 32,      lane);
        v16bf a1 = frag_from_lds(As, wm * 32 + 16, lane);
        v16bf b[4];
#pragma unroll
        for (int j = 0; j < 4; ++j) b[j] = frag_from_lds(Bs, wn * 64 + j * 16, lane);
#pragma unroll
        for (int j = 0; j < 4; ++j) {
            acc[0][j] = __builtin_amdgcn_wmma_f32_16x16x32_bf16(false, a0, false, b[j],
                                                                (short)0, acc[0][j], false, false);
            acc[1][j] = __builtin_amdgcn_wmma_f32_16x16x32_bf16(false, a1, false, b[j],
                                                                (short)0, acc[1][j], false, false);
        }
    }

    // Epilogue: out[token][col] += acc (each (token,col) owned by one block)
    const int colBase = n0 + wn * 64 + (lane & 15);
    const int rowHalf = (lane >> 4) * 8;
#pragma unroll
    for (int i = 0; i < 2; ++i) {
#pragma unroll
        for (int v = 0; v < 8; ++v) {
            const int lr = wm * 32 + i * 16 + rowHalf + v;
            const int t  = tok[lr];
            if (t >= 0) {
#pragma unroll
                for (int j = 0; j < 4; ++j) {
                    const size_t idx = (size_t)t * OUT_DIM + colBase + j * 16;
                    out[idx] += acc[i][j][v];
                }
            }
        }
    }
}

// ---------------------------------------------------------------------------
extern "C" void kernel_launch(void* const* d_in, const int* in_sizes, int n_in,
                              void* d_out, int out_size, void* d_ws, size_t ws_size,
                              hipStream_t stream) {
    const float* x       = (const float*)d_in[0];
    const float* w       = (const float*)d_in[1];
    const float* bias    = (const float*)d_in[2];
    const int*   qw      = (const int*)d_in[3];
    const int*   qz      = (const int*)d_in[4];
    const float* scales  = (const float*)d_in[5];
    const int*   indices = (const int*)d_in[6];
    float* out = (float*)d_out;

    int* offsets = (int*)d_ws;        // 16 ints (9 used)
    int* perm    = offsets + 16;      // T_TOK ints

    group_kernel<<<1, 256, 0, stream>>>(indices, offsets, perm);

    dim3 gb(OUT_DIM / BN, T_TOK / BM);
    base_gemm<<<gb, 256, 0, stream>>>(x, w, bias, out);

    dim3 gd(OUT_DIM / BN, T_TOK / BM, N_ADPT);
    delta_gemm<<<gd, 256, 0, stream>>>(x, qw, qz, scales, offsets, perm, out);
}
